// LinearAttention_54039278518801
// MI455X (gfx1250) — matrix operand order repaired and checked by reading, not compiled, not run on hardware
//
#include <hip/hip_runtime.h>
#include <cstdint>

// ---------------------------------------------------------------------------
// Types
// ---------------------------------------------------------------------------
typedef __attribute__((ext_vector_type(16))) __bf16 v16bf;
typedef __attribute__((ext_vector_type(8)))  __bf16 v8bf;
typedef __attribute__((ext_vector_type(4)))  __bf16 v4bf;
typedef __attribute__((ext_vector_type(8)))  float  v8f;
typedef __attribute__((ext_vector_type(4)))  unsigned int v4u;
typedef __attribute__((ext_vector_type(8)))  int    v8i;
typedef __attribute__((ext_vector_type(4)))  int    v4i;

#if defined(__HIP_DEVICE_COMPILE__) && defined(__gfx1250__) && \
    __has_builtin(__builtin_amdgcn_tensor_load_to_lds) &&      \
    __has_builtin(__builtin_amdgcn_s_wait_tensorcnt)
#define HAVE_TDM 1
#else
#define HAVE_TDM 0
#endif

__device__ __forceinline__ v8f wmma_bf16(v16bf a, v16bf b, v8f c) {
  // (neg_a, A, neg_b, B, c_mod, C, reuse_a, reuse_b)
  return __builtin_amdgcn_wmma_f32_16x16x32_bf16(false, a, false, b,
                                                 (short)0, c, false, false);
}

__device__ __forceinline__ v16bf cat8(v8bf lo, v8bf hi) {
  return __builtin_shufflevector(lo, hi, 0, 1, 2, 3, 4, 5, 6, 7,
                                 8, 9, 10, 11, 12, 13, 14, 15);
}

#if HAVE_TDM
// Low 32 bits of a generic pointer to __shared__ = byte offset in LDS space.
__device__ __forceinline__ uint32_t lds_off_of(const void* p) {
  return (uint32_t)(uintptr_t)p;
}

// Issue a TDM 2D tile load: tile_d0 (contiguous, elems) x tile_d1 rows of
// bf16, global row stride = stride0 elems, DMA'd into LDS at lds_off with a
// 16B LDS pad (pad_amount code 3 = 4 DW) inserted after every row:
//   pad_code 3 -> pad after 16 DW (64B row,  e.g. [..][40] layout, 80B pitch)
//   pad_code 4 -> pad after 32 DW (128B row, e.g. [..][72] layout, 144B pitch)
__device__ __forceinline__ void tdm_load_tile_bf16(const __bf16* g,
                                                   uint32_t lds_off,
                                                   uint32_t tile_d0,
                                                   uint32_t tile_d1,
                                                   uint64_t stride0,
                                                   uint32_t pad_code) {
  uint64_t ga = (uint64_t)(uintptr_t)g;
  v4u g0;
  g0[0] = 1u;                                   // count=1 (valid), user mode
  g0[1] = lds_off;                              // lds_addr (bytes)
  g0[2] = (uint32_t)ga;                         // global_addr[31:0]
  g0[3] = (uint32_t)((ga >> 32) & 0x01FFFFFFu)  // global_addr[56:32]
          | 0x80000000u;                        // type=2 ("image")
  v8i g1;
  g1[0] = (int)((1u << 16)           // data_size = 1 -> 2 bytes
                | (1u << 20)         // pad_enable
                | (pad_code << 22)   // pad_interval code
                | (3u << 25));       // pad_amount code 3 -> 4 DWORDs (16B)
  g1[1] = (int)((tile_d0 & 0xFFFFu) << 16);                 // tensor_dim0 lo
  g1[2] = (int)(((tile_d0 >> 16) & 0xFFFFu) |               // tensor_dim0 hi
                ((tile_d1 & 0xFFFFu) << 16));               // tensor_dim1 lo
  g1[3] = (int)(((tile_d1 >> 16) & 0xFFFFu) |               // tensor_dim1 hi
                ((tile_d0 & 0xFFFFu) << 16));               // tile_dim0
  g1[4] = (int)(tile_d1 & 0xFFFFu);                         // tile_dim1
  g1[5] = (int)(uint32_t)(stride0 & 0xFFFFFFFFu);           // dim0_stride lo
  g1[6] = (int)(uint32_t)((stride0 >> 32) & 0xFFFFu);       // dim0_stride hi
  g1[7] = 0;
  v4i z4 = {0, 0, 0, 0};
  v8i z8 = {0, 0, 0, 0, 0, 0, 0, 0};
  // 6-arg form (clang-23 / therock headers): (g0, g1, g2, g3, g4, cpol)
  __builtin_amdgcn_tensor_load_to_lds(g0, g1, z4, z4, z8, 0);
}
#endif  // HAVE_TDM

// ---------------------------------------------------------------------------
// fp32 -> bf16 converter (vectorized x4)
// ---------------------------------------------------------------------------
__global__ void f32_to_bf16_kernel(const float* __restrict__ in,
                                   __bf16* __restrict__ out, int n4) {
  int i = blockIdx.x * blockDim.x + threadIdx.x;
  if (i < n4) {
    float4 v = ((const float4*)in)[i];
    v4bf o;
    o[0] = (__bf16)v.x; o[1] = (__bf16)v.y;
    o[2] = (__bf16)v.z; o[3] = (__bf16)v.w;
    ((v4bf*)out)[i] = o;
  }
}

// ---------------------------------------------------------------------------
// Tiled bf16 WMMA GEMM:  C[M,N] = act( A[M,K] @ W[N,K]^T + bias[N] )
//   ACT==1: elu(x)+1  (x>0 ? x+1 : exp(x))
//   OUTF32: store fp32 to Cf, else bf16 to Cb
// Block tile 128x128, K-step 32, 256 threads = 8 waves (4 M-waves x 2 N-waves),
// each wave owns a 32x64 patch = 2x4 WMMA 16x16 accumulators.
// TDM path: double-buffered LDS; wave 0 DMAs the next K-slab (A+B tiles) via
// tensor_load_to_lds while all 8 waves run WMMA on the current slab; sync via
// s_wait_tensorcnt + workgroup barrier.
// ---------------------------------------------------------------------------
template <int ACT, bool OUTF32>
__global__ void __launch_bounds__(256)
gemm_bf16_wmma(const __bf16* __restrict__ A, const __bf16* __restrict__ W,
               const float* __restrict__ bias, __bf16* __restrict__ Cb,
               float* __restrict__ Cf, int M, int N, int K) {
  __shared__ __bf16 As[2][128][40];  // 40 = 32 + 8 pad -> 80B row (16B align)
  __shared__ __bf16 Bs[2][128][40];

  const int tid  = threadIdx.x;
  const int wid  = tid >> 5;
  const int lane = tid & 31;
  const int bm = blockIdx.y * 128;
  const int bn = blockIdx.x * 128;
  const int waveM = wid >> 1;      // 0..3
  const int waveN = wid & 1;       // 0..1

  v8f acc[2][4] = {};

  const int mrow  = lane & 15;
  const int koffA = (lane >> 4) << 3;   // 0 or 8   (A: two 8-wide K chunks)
  const int koffB = (lane >> 4) << 4;   // 0 or 16  (unused: B also reads two 8-wide chunks via koffA)

#if HAVE_TDM
  const __bf16* Abase = &A[(size_t)bm * K];
  const __bf16* Wbase = &W[(size_t)bn * K];
  if (wid == 0) {  // prologue: DMA first K-slab into buffer 0
    tdm_load_tile_bf16(Abase, lds_off_of(&As[0][0][0]), 32u, 128u,
                       (uint64_t)K, 3u);
    tdm_load_tile_bf16(Wbase, lds_off_of(&Bs[0][0][0]), 32u, 128u,
                       (uint64_t)K, 3u);
  }
  int cur = 0;
  for (int k0 = 0; k0 < K; k0 += 32) {
    if (wid == 0) __builtin_amdgcn_s_wait_tensorcnt(0);
    __syncthreads();  // buf[cur] ready for all; buf[cur^1] free to refill
    if (wid == 0 && k0 + 32 < K) {
      tdm_load_tile_bf16(Abase + k0 + 32, lds_off_of(&As[cur ^ 1][0][0]), 32u,
                         128u, (uint64_t)K, 3u);
      tdm_load_tile_bf16(Wbase + k0 + 32, lds_off_of(&Bs[cur ^ 1][0][0]), 32u,
                         128u, (uint64_t)K, 3u);
    }
#else
  const int cur = 0;
  for (int k0 = 0; k0 < K; k0 += 32) {
#pragma unroll
    for (int i = 0; i < 2; ++i) {
      int v = tid + i * 256;              // 512 vec8 per 128x32 tile
      int r = v >> 2;
      int c = (v & 3) << 3;
      *(v8bf*)&As[0][r][c] = *(const v8bf*)&A[(size_t)(bm + r) * K + k0 + c];
      *(v8bf*)&Bs[0][r][c] = *(const v8bf*)&W[(size_t)(bn + r) * K + k0 + c];
    }
    if (k0 + 32 < K) {  // prefetch next K tiles -> global_prefetch_b8
      __builtin_prefetch(&A[(size_t)(bm + (tid >> 1)) * K + k0 + 32], 0, 1);
      __builtin_prefetch(&W[(size_t)(bn + (tid >> 1)) * K + k0 + 32], 0, 1);
    }
    __syncthreads();
#endif

    v16bf afr[2], bfr[4];
#pragma unroll
    for (int t = 0; t < 2; ++t) {
      int m = waveM * 32 + t * 16 + mrow;
      v8bf lo = *(const v8bf*)&As[cur][m][koffA];
      v8bf hi = *(const v8bf*)&As[cur][m][16 + koffA];
      afr[t] = cat8(lo, hi);
    }
#pragma unroll
    for (int t = 0; t < 4; ++t) {
      int n = waveN * 64 + t * 16 + mrow;
      v8bf lo = *(const v8bf*)&Bs[cur][n][koffA];
      v8bf hi = *(const v8bf*)&Bs[cur][n][16 + koffA];
      bfr[t] = cat8(lo, hi);
    }
#pragma unroll
    for (int i = 0; i < 2; ++i)
#pragma unroll
      for (int j = 0; j < 4; ++j)
        acc[i][j] = wmma_bf16(afr[i], bfr[j], acc[i][j]);

#if HAVE_TDM
    cur ^= 1;
#else
    __syncthreads();
#endif
  }

  // Epilogue: C layout -> lanes 0-15: M=r, N=lane; lanes 16-31: M=r+8
  const int crow0 = (lane >> 4) << 3;
  const int ccol  = lane & 15;
#pragma unroll
  for (int i = 0; i < 2; ++i)
#pragma unroll
    for (int j = 0; j < 4; ++j) {
      int gn = bn + waveN * 64 + j * 16 + ccol;
      float bvl = bias[gn];
#pragma unroll
      for (int r = 0; r < 8; ++r) {
        int gm = bm + waveM * 32 + i * 16 + crow0 + r;
        float v = acc[i][j][r] + bvl;
        if (ACT == 1) v = (v > 0.0f) ? (v + 1.0f) : __expf(v);
        size_t idx = (size_t)gm * N + gn;
        if (OUTF32) Cf[idx] = v;
        else        Cb[idx] = (__bf16)v;
      }
    }
}

// ---------------------------------------------------------------------------
// K_sum[b,h,d] = sum_s K[b,s,h,d]     (K bf16 laid out [B,S,H*D])
// ---------------------------------------------------------------------------
__global__ void __launch_bounds__(256)
ksum_kernel(const __bf16* __restrict__ Kb, float* __restrict__ Ksum) {
  __shared__ float red[256];
  const int bh = blockIdx.x;              // b*16 + h
  const int b = bh >> 4, h = bh & 15;
  const int tid = threadIdx.x;
  const int d = tid & 63, g = tid >> 6;   // 4 s-groups
  const size_t base = ((size_t)b * 4096) * 1024 + h * 64 + d;
  float s = 0.0f;
  for (int si = g; si < 4096; si += 4)
    s += (float)Kb[base + (size_t)si * 1024];
  red[tid] = s;
  __syncthreads();
  if (tid < 64)
    Ksum[bh * 64 + tid] =
        red[tid] + red[tid + 64] + red[tid + 128] + red[tid + 192];
}

// ---------------------------------------------------------------------------
// KV[b,h,d,e] = sum_s K[b,s,h,d] * V[b,s,h,e]   via WMMA (A = K^T)
// One block per (b,h); 8 waves, each owns 2 of the 16 16x16 output tiles.
// TDM path: double-buffered 32-row s-slabs of K and V DMA'd into LDS while
// WMMA consumes the previous slab.
// ---------------------------------------------------------------------------
__global__ void __launch_bounds__(256)
kv_wmma_kernel(const __bf16* __restrict__ Kb, const __bf16* __restrict__ Vb,
               float* __restrict__ KV) {
#if HAVE_TDM
  __shared__ __bf16 Ks[2][32][72];  // [s][d], 144B pitch (16B aligned)
  __shared__ __bf16 Vs[2][32][72];  // [s][e]
#else
  __shared__ __bf16 Ks[1][32][72];
  __shared__ __bf16 Vs[1][32][72];
#endif
  const int bh = blockIdx.x;
  const int b = bh >> 4, h = bh & 15;
  const int tid = threadIdx.x, wid = tid >> 5, lane = tid & 31;
  const int dtile = wid >> 1;          // 0..3  (d rows of KV)
  const int e0 = (wid & 1) * 32;       // two e tiles: e0, e0+16

  v8f acc[2] = {};
  const size_t base = ((size_t)b * 4096) * 1024 + h * 64;
  const int mrow = lane & 15;
  const int koffA = (lane >> 4) << 3;
  const int koffB = (lane >> 4) << 4;

#if HAVE_TDM
  const __bf16* Kbase = &Kb[base];
  const __bf16* Vbase = &Vb[base];
  if (wid == 0) {  // prologue: first 32-row slab into buffer 0
    tdm_load_tile_bf16(Kbase, lds_off_of(&Ks[0][0][0]), 64u, 32u, 1024u, 4u);
    tdm_load_tile_bf16(Vbase, lds_off_of(&Vs[0][0][0]), 64u, 32u, 1024u, 4u);
  }
  int cur = 0;
#else
  const int cur = 0;
  const int srow = tid >> 3, dcol = (tid & 7) << 3;
#endif

  for (int s0 = 0; s0 < 4096; s0 += 32) {
#if HAVE_TDM
    if (wid == 0) __builtin_amdgcn_s_wait_tensorcnt(0);
    __syncthreads();  // buf[cur] ready; buf[cur^1] free to refill
    if (wid == 0 && s0 + 32 < 4096) {
      tdm_load_tile_bf16(Kbase + (size_t)(s0 + 32) * 1024,
                         lds_off_of(&Ks[cur ^ 1][0][0]), 64u, 32u, 1024u, 4u);
      tdm_load_tile_bf16(Vbase + (size_t)(s0 + 32) * 1024,
                         lds_off_of(&Vs[cur ^ 1][0][0]), 64u, 32u, 1024u, 4u);
    }
#else
    *(v8bf*)&Ks[0][srow][dcol] =
        *(const v8bf*)&Kb[base + (size_t)(s0 + srow) * 1024 + dcol];
    *(v8bf*)&Vs[0][srow][dcol] =
        *(const v8bf*)&Vb[base + (size_t)(s0 + srow) * 1024 + dcol];
    __syncthreads();
#endif

    v16bf afr;          // A = K^T: (m=d, k=s) -> transposed LDS read
    const int m = dtile * 16 + mrow;
#pragma unroll
    for (int e = 0; e < 16; ++e) {
      int k = koffB + e;
      afr[e] = Ks[cur][k][m];
    }
#pragma unroll
    for (int t = 0; t < 2; ++t) {
      v16bf bfr;        // B = V: (k=s, n=e)
#pragma unroll
      for (int e = 0; e < 16; ++e)
        bfr[e] = Vs[cur][koffB + e][e0 + t * 16 + mrow];
      acc[t] = wmma_bf16(afr, bfr, acc[t]);
    }

#if HAVE_TDM
    cur ^= 1;
#else
    __syncthreads();
#endif
  }

  const int crow0 = (lane >> 4) << 3, ccol = lane & 15;
#pragma unroll
  for (int t = 0; t < 2; ++t)
#pragma unroll
    for (int r = 0; r < 8; ++r) {
      int d = dtile * 16 + crow0 + r;
      int e = e0 + t * 16 + ccol;
      KV[((size_t)bh * 64 + d) * 64 + e] = acc[t][r];
    }
}

// ---------------------------------------------------------------------------
// att[b,s,h,e] = Z(b,s,h) * sum_d Q[b,s,h,d] * KV[b,h,d,e]
// One block per (b,h, 128-row s-chunk). Z computed in-block from Ksum.
// TDM path: Q tile (128x64) and KV tile (64x64) DMA'd into LDS by wave 0.
// ---------------------------------------------------------------------------
__global__ void __launch_bounds__(256)
apply_attn_kernel(const __bf16* __restrict__ Qb, const __bf16* __restrict__ KVb,
                  const float* __restrict__ Ksum, __bf16* __restrict__ attb) {
  __shared__ __bf16 Qs[128][72];
  __shared__ __bf16 KVs[64][72];
  __shared__ float zsh[128];
  __shared__ float ksl[64];

  const int blk = blockIdx.x;
  const int bh = blk >> 5;            // (b*16+h)
  const int schunk = blk & 31;
  const int b = bh >> 4, h = bh & 15;
  const int s0 = schunk * 128;
  const int tid = threadIdx.x, wid = tid >> 5, lane = tid & 31;

  const size_t qbase = ((size_t)b * 4096 + s0) * 1024 + h * 64;
  const size_t kvb = (size_t)bh * 4096;

#if HAVE_TDM
  if (wid == 0) {
    tdm_load_tile_bf16(&Qb[qbase], lds_off_of(&Qs[0][0]), 64u, 128u, 1024u, 4u);
    tdm_load_tile_bf16(&KVb[kvb], lds_off_of(&KVs[0][0]), 64u, 64u, 64u, 4u);
  }
  if (tid < 64) ksl[tid] = Ksum[bh * 64 + tid];
  if (wid == 0) __builtin_amdgcn_s_wait_tensorcnt(0);
  __syncthreads();
#else
#pragma unroll
  for (int i = 0; i < 4; ++i) {       // stage Q 128x64
    int v = tid + i * 256;
    int r = v >> 3, c = (v & 7) << 3;
    *(v8bf*)&Qs[r][c] = *(const v8bf*)&Qb[qbase + (size_t)r * 1024 + c];
  }
#pragma unroll
  for (int i = 0; i < 2; ++i) {       // stage KV 64x64
    int v = tid + i * 256;
    int r = v >> 3, c = (v & 7) << 3;
    *(v8bf*)&KVs[r][c] = *(const v8bf*)&KVb[kvb + r * 64 + c];
  }
  if (tid < 64) ksl[tid] = Ksum[bh * 64 + tid];
  __syncthreads();
#endif

  if (tid < 128) {                    // normalizer Z per row
    float a = 0.0f;
    for (int d = 0; d < 64; ++d) a += (float)Qs[tid][d] * ksl[d];
    zsh[tid] = 1.0f / (a + 1e-6f);
  }
  __syncthreads();

  // wave 'wid' computes rows wid*16..wid*16+15, all 64 cols (4 tiles), K=64
  v8f acc[4] = {};
  const int mrow = lane & 15;
  const int koffA = (lane >> 4) << 3;
  const int koffB = (lane >> 4) << 4;
#pragma unroll
  for (int ks = 0; ks < 2; ++ks) {
    const int kb = ks * 32;
    v16bf afr;
    {
      int m = wid * 16 + mrow;
      v8bf lo = *(const v8bf*)&Qs[m][kb + koffB];
      v8bf hi = *(const v8bf*)&Qs[m][kb + koffB + 8];
      afr = cat8(lo, hi);
    }
#pragma unroll
    for (int t = 0; t < 4; ++t) {
      v16bf bfr;
#pragma unroll
      for (int e = 0; e < 16; ++e)
        bfr[e] = KVs[kb + koffB + e][t * 16 + mrow];
      acc[t] = wmma_bf16(afr, bfr, acc[t]);
    }
  }

  const int crow0 = (lane >> 4) << 3, ccol = lane & 15;
  const size_t obase = qbase;
#pragma unroll
  for (int t = 0; t < 4; ++t)
#pragma unroll
    for (int r = 0; r < 8; ++r) {
      int row = wid * 16 + crow0 + r;
      int col = t * 16 + ccol;
      float v = acc[t][r] * zsh[row];
      attb[obase + (size_t)row * 1024 + col] = (__bf16)v;
    }
}

// ---------------------------------------------------------------------------
// Host launch
// ---------------------------------------------------------------------------
extern "C" void kernel_launch(void* const* d_in, const int* in_sizes, int n_in,
                              void* d_out, int out_size, void* d_ws,
                              size_t ws_size, hipStream_t stream) {
  const float* x  = (const float*)d_in[0];
  const float* Wq = (const float*)d_in[1];
  const float* bq = (const float*)d_in[2];
  const float* Wk = (const float*)d_in[3];
  const float* bk = (const float*)d_in[4];
  const float* Wv = (const float*)d_in[5];
  const float* bv = (const float*)d_in[6];
  const float* Wo = (const float*)d_in[7];
  const float* bo = (const float*)d_in[8];
  float* out = (float*)d_out;

  const int Bc = 4, Sc = 4096, Ec = 1024, Hc = 16;
  const int M = Bc * Sc;  // 16384

  char* base = (char*)d_ws;
  size_t off = 0;
  auto take = [&](size_t bytes) -> char* {
    char* p = base + off;
    off += (bytes + 255) & ~(size_t)255;
    return p;
  };

  __bf16* xb   = (__bf16*)take((size_t)M * Ec * 2);
  __bf16* Wqb  = (__bf16*)take((size_t)Ec * Ec * 2);
  __bf16* Wkb  = (__bf16*)take((size_t)Ec * Ec * 2);
  __bf16* Wvb  = (__bf16*)take((size_t)Ec * Ec * 2);
  __bf16* Wob  = (__bf16*)take((size_t)Ec * Ec * 2);
  __bf16* Qbuf = (__bf16*)take((size_t)M * Ec * 2);
  __bf16* Kbuf = (__bf16*)take((size_t)M * Ec * 2);
  __bf16* Vbuf = (__bf16*)take((size_t)M * Ec * 2);
  __bf16* Abuf = (__bf16*)take((size_t)M * Ec * 2);   // attention output
  float*  KV   = (float*)take((size_t)Bc * Hc * 64 * 64 * 4);
  __bf16* KVb  = (__bf16*)take((size_t)Bc * Hc * 64 * 64 * 2);
  float*  Ksum = (float*)take((size_t)Bc * Hc * 64 * 4);
  (void)ws_size; (void)in_sizes; (void)n_in; (void)out_size;

  // 1) fp32 -> bf16 conversions
  {
    int n4 = (M * Ec) / 4;
    f32_to_bf16_kernel<<<(n4 + 255) / 256, 256, 0, stream>>>(x, xb, n4);
    int w4 = (Ec * Ec) / 4;
    f32_to_bf16_kernel<<<(w4 + 255) / 256, 256, 0, stream>>>(Wq, Wqb, w4);
    f32_to_bf16_kernel<<<(w4 + 255) / 256, 256, 0, stream>>>(Wk, Wkb, w4);
    f32_to_bf16_kernel<<<(w4 + 255) / 256, 256, 0, stream>>>(Wv, Wvb, w4);
    f32_to_bf16_kernel<<<(w4 + 255) / 256, 256, 0, stream>>>(Wo, Wob, w4);
  }

  // 2) projections (Q,K with elu+1 epilogue)
  dim3 ggrid(Ec / 128, M / 128);
  gemm_bf16_wmma<1, false><<<ggrid, 256, 0, stream>>>(xb, Wqb, bq, Qbuf,
                                                      nullptr, M, Ec, Ec);
  gemm_bf16_wmma<1, false><<<ggrid, 256, 0, stream>>>(xb, Wkb, bk, Kbuf,
                                                      nullptr, M, Ec, Ec);
  gemm_bf16_wmma<0, false><<<ggrid, 256, 0, stream>>>(xb, Wvb, bv, Vbuf,
                                                      nullptr, M, Ec, Ec);

  // 3) K_sum and KV summary
  ksum_kernel<<<Bc * Hc, 256, 0, stream>>>(Kbuf, Ksum);
  kv_wmma_kernel<<<Bc * Hc, 256, 0, stream>>>(Kbuf, Vbuf, KV);
  {
    int n4 = (Bc * Hc * 64 * 64) / 4;
    f32_to_bf16_kernel<<<(n4 + 255) / 256, 256, 0, stream>>>(KV, KVb, n4);
  }

  // 4) att = Z * (Q @ KV)
  apply_attn_kernel<<<Bc * Hc * (Sc / 128), 256, 0, stream>>>(Qbuf, KVb, Ksum,
                                                              Abuf);

  // 5) out = att @ Wo^T + bo   (fp32 to d_out)
  gemm_bf16_wmma<0, true><<<ggrid, 256, 0, stream>>>(Abuf, Wob, bo, nullptr,
                                                     out, M, Ec, Ec);
}
